// DeltaNetAttention_16535624090013
// MI455X (gfx1250) — matrix operand order repaired and checked by reading, not compiled, not run on hardware
//
#include <hip/hip_runtime.h>
#include <hip/hip_bf16.h>
#include <math.h>

typedef __attribute__((ext_vector_type(16))) _Float16 v16h;
typedef __attribute__((ext_vector_type(8)))  _Float16 v8h;
typedef __attribute__((ext_vector_type(8)))  float    v8f;

#define HIDDEN_ 1024
#define HEADS_  16
#define HEADD_  64
#define BATCH_  4
#define SEQT_   1024

// ---------------------------------------------------------------------------
// fp32 -> f16 conversion (feeds the WMMA path)
// ---------------------------------------------------------------------------
__global__ void cvt_f32_to_f16(const float* __restrict__ in,
                               _Float16* __restrict__ out, int n) {
    int i = blockIdx.x * blockDim.x + threadIdx.x;
    if (i < n) out[i] = (_Float16)in[i];
}

// ---------------------------------------------------------------------------
// C[M,N] = A[M,K] (f16, row-major) x W[N,K]^T (f16, row-major) + bias[N]
// One wave computes a 16(M) x 64(N) strip: A-fragment reused across 4 WMMAs.
// Block = 128 threads (4 waves) -> 64x64 tile. Grid = (M/64, N/64).
// A fragment layout (16-bit 16x32): lane m (lo 4 bits) holds row m; the lane's
// upper bit (hi) selects contiguous K = hi*16..+16.
// B fragment layout (16-bit 32x16): lane holds column n = lane&15, contiguous
// K chunk of 16 starting at kk + hi*16.
// ---------------------------------------------------------------------------
__global__ void gemm_f16_wmma(const _Float16* __restrict__ A,
                              const _Float16* __restrict__ W,
                              const float* __restrict__ bias,
                              float* __restrict__ C,
                              int M, int N, int K) {
    const int lane = threadIdx.x & 31;
    const int wave = threadIdx.x >> 5;
    const int lo16 = lane & 15;
    const int hi   = lane >> 4;

    const int mTile = blockIdx.x * 64 + wave * 16;
    const int nBase = blockIdx.y * 64;

    const _Float16* aRow = A + (size_t)(mTile + lo16) * K;

    v8f acc[4] = {v8f{}, v8f{}, v8f{}, v8f{}};

    for (int kk = 0; kk < K; kk += 32) {
        v8h a0 = *(const v8h*)(aRow + kk + hi * 16);
        v8h a1 = *(const v8h*)(aRow + kk + hi * 16 + 8);
        v16h a;
#pragma unroll
        for (int i = 0; i < 8; ++i) { a[i] = a0[i]; a[8 + i] = a1[i]; }
#pragma unroll
        for (int t = 0; t < 4; ++t) {
            const int n = nBase + t * 16 + lo16;
            v16h b = *(const v16h*)(W + (size_t)n * K + kk + hi * 16);
            acc[t] = __builtin_amdgcn_wmma_f32_16x16x32_f16(
                false, a, false, b, (short)0, acc[t], false, false);
        }
    }

#pragma unroll
    for (int t = 0; t < 4; ++t) {
        const int n = nBase + t * 16 + lo16;
        const float bn = bias[n];
#pragma unroll
        for (int r = 0; r < 8; ++r) {
            const int m = mTile + hi * 8 + r;           // C layout: M = hi*8 + r
            C[(size_t)m * N + n] = acc[t][r] + bn;
        }
    }
}

// ---------------------------------------------------------------------------
// L2-normalize contiguous 64-element vectors in place: x / max(||x||, 1e-12).
// One wave32 per vector; lane holds elements (lane, lane+32).
// ---------------------------------------------------------------------------
__global__ void l2norm64(float* __restrict__ x, int nvec) {
    const int wid  = (blockIdx.x * blockDim.x + threadIdx.x) >> 5;
    const int lane = threadIdx.x & 31;
    if (wid >= nvec) return;
    float* v = x + (size_t)wid * 64;
    float a = v[lane];
    float b = v[lane + 32];
    float ss = a * a + b * b;
#pragma unroll
    for (int off = 16; off; off >>= 1) ss += __shfl_xor(ss, off, 32);
    const float inv = 1.0f / fmaxf(sqrtf(ss), 1e-12f);
    v[lane]      = a * inv;
    v[lane + 32] = b * inv;
}

// ---------------------------------------------------------------------------
// Sequential DeltaNet scan. One block (64 threads) per (b,h) pair.
// S lives in LDS with pitch 65 -> conflict-free column AND row access.
// Thread e owns state column e (v_old reduction, rank-1 update, rescale);
// o_i row reduction crosses columns and is fenced by barriers.
// ---------------------------------------------------------------------------
__global__ void deltanet_scan(const float* __restrict__ Q,
                              const float* __restrict__ Kb,
                              const float* __restrict__ V,
                              float* __restrict__ O,
                              float* __restrict__ stateOut) {
    const int pair = blockIdx.x;           // b*HEADS + h
    const int b = pair >> 4;
    const int h = pair & 15;
    const int e = threadIdx.x;             // 0..63: owned state column

    __shared__ float S[64 * 65];
    __shared__ float ks[64];
    __shared__ float qs[64];
    __shared__ float red[2];

    for (int d = 0; d < 64; ++d) S[d * 65 + e] = 0.0f;
    __syncthreads();

    const int col = h * HEADD_;
    for (int t = 0; t < SEQT_; ++t) {
        const size_t row = (size_t)(b * SEQT_ + t) * HIDDEN_ + col;
        const float kv = Kb[row + e];
        const float qv = Q[row + e];
        const float vv = V[row + e];
        ks[e] = kv;
        qs[e] = qv;
        __syncthreads();

        // v_old[e] = sum_d k[d] * S[d][e]   (own column)
        float vo = 0.0f;
        for (int d = 0; d < 64; ++d) vo = fmaf(ks[d], S[d * 65 + e], vo);
        const float delta = fminf(fmaxf(vv - vo, -1.0f), 1.0f);

        // S[:,e] += clip(beta * k * delta); exact column max for free
        float cmax = 0.0f;
        for (int d = 0; d < 64; ++d) {
            const float u = fminf(fmaxf(0.9f * ks[d] * delta, -1.0f), 1.0f);
            const float s = S[d * 65 + e] + u;
            S[d * 65 + e] = s;
            cmax = fmaxf(cmax, fabsf(s));
        }
#pragma unroll
        for (int off = 16; off; off >>= 1)
            cmax = fmaxf(cmax, __shfl_xor(cmax, off, 32));
        if ((e & 31) == 0) red[e >> 5] = cmax;
        __syncthreads();
        const float m = fmaxf(red[0], red[1]);
        if (m > 10.0f) {
            for (int d = 0; d < 64; ++d) S[d * 65 + e] *= 0.9f;
        }
        __syncthreads();

        // o[e] = clip(sum_j S[e][j] * q[j])   (row read, conflict-free pitch)
        float o = 0.0f;
        for (int j = 0; j < 64; ++j) o = fmaf(S[e * 65 + j], qs[j], o);
        O[row + e] = fminf(fmaxf(o, -1.0f), 1.0f);
        __syncthreads();
    }

    // final_state[b,h,d,e]
    for (int d = 0; d < 64; ++d)
        stateOut[((size_t)pair * 64 + d) * 64 + e] = S[d * 65 + e];
}

// ---------------------------------------------------------------------------
extern "C" void kernel_launch(void* const* d_in, const int* in_sizes, int n_in,
                              void* d_out, int out_size, void* d_ws, size_t ws_size,
                              hipStream_t stream) {
    (void)in_sizes; (void)n_in; (void)out_size; (void)ws_size;

    const float* X  = (const float*)d_in[0];
    const float* Wq = (const float*)d_in[1];
    const float* bq = (const float*)d_in[2];
    const float* Wk = (const float*)d_in[3];
    const float* bk = (const float*)d_in[4];
    const float* Wv = (const float*)d_in[5];
    const float* bv = (const float*)d_in[6];
    const float* Wo = (const float*)d_in[7];
    const float* bo = (const float*)d_in[8];

    const int M  = BATCH_ * SEQT_;          // 4096
    const int NK = HIDDEN_;                 // 1024
    const int nX = M * HIDDEN_;             // 4,194,304
    const int nW = HIDDEN_ * HIDDEN_;       // 1,048,576

    // workspace carve-up
    char* ws = (char*)d_ws;
    size_t off = 0;
    _Float16* Xh  = (_Float16*)(ws + off); off += (size_t)nX * sizeof(_Float16); // 8 MB
    _Float16* Wqh = (_Float16*)(ws + off); off += (size_t)nW * sizeof(_Float16);
    _Float16* Wkh = (_Float16*)(ws + off); off += (size_t)nW * sizeof(_Float16);
    _Float16* Wvh = (_Float16*)(ws + off); off += (size_t)nW * sizeof(_Float16);
    _Float16* Woh = (_Float16*)(ws + off); off += (size_t)nW * sizeof(_Float16);
    float* Qf = (float*)(ws + off); off += (size_t)nX * sizeof(float);
    float* Kf = (float*)(ws + off); off += (size_t)nX * sizeof(float);
    float* Vf = (float*)(ws + off); off += (size_t)nX * sizeof(float);
    float* Of = (float*)(ws + off); off += (size_t)nX * sizeof(float);
    _Float16* Oh = Xh;   // X no longer needed once Q/K/V GEMMs are done

    float* out   = (float*)d_out;
    float* state = out + (size_t)nX;        // final_state after [B,T,HIDDEN]

    // 1) convert inputs to f16
    cvt_f32_to_f16<<<(nX + 255) / 256, 256, 0, stream>>>(X,  Xh,  nX);
    cvt_f32_to_f16<<<(nW + 255) / 256, 256, 0, stream>>>(Wq, Wqh, nW);
    cvt_f32_to_f16<<<(nW + 255) / 256, 256, 0, stream>>>(Wk, Wkh, nW);
    cvt_f32_to_f16<<<(nW + 255) / 256, 256, 0, stream>>>(Wv, Wvh, nW);
    cvt_f32_to_f16<<<(nW + 255) / 256, 256, 0, stream>>>(Wo, Woh, nW);

    // 2) Q/K/V projections (WMMA)
    dim3 gGemm(M / 64, NK / 64);            // (64, 16)
    gemm_f16_wmma<<<gGemm, 128, 0, stream>>>(Xh, Wqh, bq, Qf, M, NK, NK);
    gemm_f16_wmma<<<gGemm, 128, 0, stream>>>(Xh, Wkh, bk, Kf, M, NK, NK);
    gemm_f16_wmma<<<gGemm, 128, 0, stream>>>(Xh, Wvh, bv, Vf, M, NK, NK);

    // 3) l2norm on q, k (one wave per 64-vector)
    const int nvec = M * HEADS_;            // 65536
    const int nrmBlocks = (nvec * 32 + 255) / 256;
    l2norm64<<<nrmBlocks, 256, 0, stream>>>(Qf, nvec);
    l2norm64<<<nrmBlocks, 256, 0, stream>>>(Kf, nvec);

    // 4) sequential scan: one block per (b,h)
    deltanet_scan<<<BATCH_ * HEADS_, 64, 0, stream>>>(Qf, Kf, Vf, Of, state);

    // 5) output projection (WMMA) into d_out
    cvt_f32_to_f16<<<(nX + 255) / 256, 256, 0, stream>>>(Of, Oh, nX);
    gemm_f16_wmma<<<gGemm, 128, 0, stream>>>(Oh, Woh, bo, out, M, NK, NK);
}
